// VAE_CDDD_decoder_20358144983638
// MI455X (gfx1250) — matrix operand order repaired and checked by reading, not compiled, not run on hardware
//
#include <hip/hip_runtime.h>

// ---------------- CDNA5 types ----------------
typedef __attribute__((ext_vector_type(16))) __bf16 v16bf;
typedef __attribute__((ext_vector_type(8)))  float  v8f;
typedef __attribute__((ext_vector_type(4)))  float  v4f;

// ---------------- problem constants ----------------
#define B_TOT  1024
#define SEQ    350
#define NVOCAB 41
#define SOS_TOK 1
#define EMB    512
#define H1     256
#define H2     128
#define H3     32
#define G1     (3*H1)   // 768
#define G2     (3*H2)   // 384
#define G3     (3*H3)   // 96
#define NPROJ  48       // 41 padded to 3 WMMA column tiles

#define M_WG    32                  // batch rows per workgroup (2 row tiles)
#define NWG     (B_TOT / M_WG)      // 32 persistent workgroups
#define THREADS 256
#define WAVES   8

// fragment-packed weight sizes (elements): NT*KT*512
#define FR_WHH0 (48*8*512)   // 196608
#define FR_WIH1 (24*8*512)   // 98304
#define FR_WHH1 (24*4*512)   // 49152
#define FR_WIH2 ( 6*4*512)   // 12288
#define FR_WHH2 ( 6*1*512)   // 3072
#define FR_PROJ ( 3*1*512)   // 1536

// =====================================================================
// Prep kernel 1: repack W (Nact x K, row-major f32) into WMMA-B
// fragment-major bf16.  Tile (nt,kt): 32 lanes x 16 bf16 contiguous.
// Element j of lane L maps to  n = nt*16 + (L&15),  k = kt*32 + (L>>4)*16 + j
// (contiguous k order; the A loads in gemm_cols use the same).  n >= Nact pads with 0.
// =====================================================================
__global__ void pack_frag_kernel(const float* __restrict__ W, __bf16* __restrict__ dst,
                                 int Nact, int K, int NT, int KT) {
  int idx = blockIdx.x * blockDim.x + threadIdx.x;
  int total = NT * KT * 512;
  if (idx >= total) return;
  int tile = idx >> 9;
  int r    = idx & 511;
  int lane = r >> 4;
  int j    = r & 15;
  int nt = tile / KT;
  int kt = tile - nt * KT;
  int n  = nt * 16 + (lane & 15);
  int k  = kt * 32 + (lane >> 4) * 16 + j;
  float v = (n < Nact) ? W[(size_t)n * K + k] : 0.0f;
  dst[idx] = (__bf16)v;
}

// =====================================================================
// Prep kernel 2: gi0 table (41 x 768):
//   tab[v][j] = sum_k (W_emb[k][v] + b_emb[k]) * Wih0[j][k] + bih0[j]
// =====================================================================
__global__ void gi0_table_kernel(const float* __restrict__ Wemb, const float* __restrict__ bemb,
                                 const float* __restrict__ Wih0, const float* __restrict__ bih0,
                                 float* __restrict__ tab) {
  int idx = blockIdx.x * blockDim.x + threadIdx.x;
  if (idx >= NVOCAB * G1) return;
  int v = idx / G1;
  int j = idx - v * G1;
  float s = bih0[j];
  const float* wr = Wih0 + (size_t)j * EMB;
  for (int k = 0; k < EMB; ++k)
    s += (Wemb[(size_t)k * NVOCAB + v] + bemb[k]) * wr[k];
  tab[idx] = s;
}

// =====================================================================
// Prep kernel 3: initial hiddens: combined = latent @ W_init^T + b_init
// split into h0 (256), h1 (128), h2 (32) per batch row.
// =====================================================================
__global__ void init_hidden_kernel(const float* __restrict__ latent,
                                   const float* __restrict__ Winit, const float* __restrict__ binit,
                                   float* __restrict__ h0i, float* __restrict__ h1i,
                                   float* __restrict__ h2i) {
  int idx = blockIdx.x * blockDim.x + threadIdx.x;
  if (idx >= B_TOT * (H1 + H2 + H3)) return;
  int b = idx / (H1 + H2 + H3);
  int o = idx - b * (H1 + H2 + H3);
  const float* lr = latent + (size_t)b * EMB;
  const float* wr = Winit + (size_t)o * EMB;
  float s = binit[o];
  for (int k = 0; k < EMB; ++k) s += lr[k] * wr[k];
  if (o < H1)            h0i[(size_t)b * H1 + o] = s;
  else if (o < H1 + H2)  h1i[(size_t)b * H2 + (o - H1)] = s;
  else                   h2i[(size_t)b * H3 + (o - H1 - H2)] = s;
}

// =====================================================================
// Wave-level WMMA GEMM, column-tile ownership with B-fragment reuse:
// each wave owns column tiles nt (round-robin, scalar loop bound via
// readfirstlane so EXEC stays all-ones at every v_wmma), computes BOTH
// row tiles (mt=0,1) with two accumulators so every B fragment is
// fetched from L2 exactly once.  A fragments (invariant across nt) are
// hoisted into registers up front.
// =====================================================================
template<int KT, int NT>
__device__ __forceinline__ void gemm_cols(const __bf16* __restrict__ aLds, int lda,
                                          const __bf16* __restrict__ wfrag,
                                          float* __restrict__ outLds, int ldo,
                                          int lane, int wv) {
  const int g  = lane >> 4;
  const int lr = lane & 15;
  union F { v16bf v; v4f q[2]; };

  // ---- hoist A fragments for both row tiles (contiguous k order, as in the B packing:
  //      lane half g holds K = {16g..16g+15} per 32-K tile) ----
  F A0[KT], A1[KT];
  const __bf16* ar0 = aLds + (size_t)lr * lda;          // rows 0..15
  const __bf16* ar1 = aLds + (size_t)(16 + lr) * lda;   // rows 16..31
#pragma unroll
  for (int kt = 0; kt < KT; ++kt) {
    const int base = kt * 32 + g * 16;
    A0[kt].q[0] = *reinterpret_cast<const v4f*>(ar0 + base);
    A0[kt].q[1] = *reinterpret_cast<const v4f*>(ar0 + base + 8);
    A1[kt].q[0] = *reinterpret_cast<const v4f*>(ar1 + base);
    A1[kt].q[1] = *reinterpret_cast<const v4f*>(ar1 + base + 8);
  }

  for (int nt = wv; nt < NT; nt += WAVES) {
    v8f acc0 = {0.f, 0.f, 0.f, 0.f, 0.f, 0.f, 0.f, 0.f};
    v8f acc1 = {0.f, 0.f, 0.f, 0.f, 0.f, 0.f, 0.f, 0.f};
    const __bf16* b = wfrag + (size_t)nt * KT * 512 + lane * 16;
#pragma unroll
    for (int kt = 0; kt < KT; ++kt) {
      F Bm;
      Bm.q[0] = *reinterpret_cast<const v4f*>(b);
      Bm.q[1] = *reinterpret_cast<const v4f*>(b + 8);
      __builtin_prefetch(b + 512, 0, 1);     // stream next fragment (global_prefetch_b8)
      acc0 = __builtin_amdgcn_wmma_f32_16x16x32_bf16(false, A0[kt].v, false, Bm.v,
                                                     (short)0, acc0, false, false);
      acc1 = __builtin_amdgcn_wmma_f32_16x16x32_bf16(false, A1[kt].v, false, Bm.v,
                                                     (short)0, acc1, false, false);
      b += 512;
    }
    // C/D layout: VGPR r -> m = 8g + r (tile 0) / 16 + 8g + r (tile 1), n = nt*16 + lr
    float* o0 = outLds + (size_t)(g * 8) * ldo + nt * 16 + lr;
    float* o1 = outLds + (size_t)(16 + g * 8) * ldo + nt * 16 + lr;
#pragma unroll
    for (int r = 0; r < 8; ++r) {
      o0[(size_t)r * ldo] = acc0[r];
      o1[(size_t)r * ldo] = acc1[r];
    }
  }
}

__device__ __forceinline__ float sigmoidf_(float x) {
  return 1.0f / (1.0f + __expf(-x));
}

// =====================================================================
// Persistent GRU kernel: each workgroup owns 32 batch rows for all 350
// steps.  Hidden state lives in LDS (f32 master + bf16 WMMA-A copy).
// Weights stream from L2 as bf16 fragments (each read exactly once/WG/step).
// =====================================================================
__global__ void __launch_bounds__(THREADS)
gru_persistent_kernel(const int*   __restrict__ tgt,
                      const float* __restrict__ gi0tab,
                      const float* __restrict__ h0i,
                      const float* __restrict__ h1i,
                      const float* __restrict__ h2i,
                      const __bf16* __restrict__ fWhh0,
                      const __bf16* __restrict__ fWih1,
                      const __bf16* __restrict__ fWhh1,
                      const __bf16* __restrict__ fWih2,
                      const __bf16* __restrict__ fWhh2,
                      const __bf16* __restrict__ fProj,
                      const float* __restrict__ bhh0,
                      const float* __restrict__ bih1,
                      const float* __restrict__ bhh1,
                      const float* __restrict__ bih2,
                      const float* __restrict__ bhh2,
                      const float* __restrict__ bproj,
                      float* __restrict__ outLogits,
                      float* __restrict__ outPred) {
  extern __shared__ char smemRaw[];
  float*  h0f  = (float*)smemRaw;                 // 32*256 f32
  float*  h1f  = h0f + M_WG * H1;                 // 32*128 f32
  float*  h2f  = h1f + M_WG * H2;                 // 32*32  f32
  float*  gbuf = h2f + M_WG * H3;                 // 32*768 f32 (reused per layer)
  __bf16* h0b  = (__bf16*)(gbuf + M_WG * G1);     // 32*256 bf16
  __bf16* h1b  = h0b + M_WG * H1;                 // 32*128 bf16
  __bf16* h2b  = h1b + M_WG * H2;                 // 32*32  bf16
  int*    tokL = (int*)(h2b + M_WG * H3);         // 32 ints

  const int tid  = threadIdx.x;
  const int lane = tid & 31;
  const int wv   = __builtin_amdgcn_readfirstlane(tid >> 5);  // scalar wave id
  const int bm0  = blockIdx.x * M_WG;

  // ---- load initial hidden state into LDS ----
  for (int i = tid; i < M_WG * H1; i += THREADS) {
    int m = i >> 8, j = i & (H1 - 1);
    float v = h0i[(size_t)(bm0 + m) * H1 + j];
    h0f[i] = v; h0b[i] = (__bf16)v;
  }
  for (int i = tid; i < M_WG * H2; i += THREADS) {
    int m = i >> 7, j = i & (H2 - 1);
    float v = h1i[(size_t)(bm0 + m) * H2 + j];
    h1f[i] = v; h1b[i] = (__bf16)v;
  }
  for (int i = tid; i < M_WG * H3; i += THREADS) {
    int m = i >> 5, j = i & (H3 - 1);
    float v = h2i[(size_t)(bm0 + m) * H3 + j];
    h2f[i] = v; h2b[i] = (__bf16)v;
  }

  for (int t = 0; t < SEQ; ++t) {
    __syncthreads();   // gbuf reuse + initial-state visibility
    if (tid < M_WG)
      tokL[tid] = (t == 0) ? SOS_TOK : tgt[(size_t)(bm0 + tid) * SEQ + t];

    // ---- layer 0: gh0 = h0 @ Whh0^T  (K=256 -> N=768) ----
    gemm_cols<8, 48>(h0b, H1, fWhh0, gbuf, G1, lane, wv);
    __syncthreads();
    for (int i = tid; i < M_WG * H1; i += THREADS) {
      int m = i >> 8, j = i & (H1 - 1);
      const float* gi = gi0tab + (size_t)tokL[m] * G1;   // includes bih0
      float hr = gbuf[(size_t)m * G1 + j]          + bhh0[j];
      float hz = gbuf[(size_t)m * G1 + j + H1]     + bhh0[j + H1];
      float hn = gbuf[(size_t)m * G1 + j + 2 * H1] + bhh0[j + 2 * H1];
      float r = sigmoidf_(gi[j] + hr);
      float z = sigmoidf_(gi[j + H1] + hz);
      float n = tanhf(gi[j + 2 * H1] + r * hn);
      float h = (1.0f - z) * n + z * h0f[i];
      h0f[i] = h; h0b[i] = (__bf16)h;
    }
    __syncthreads();

    // ---- layer 1: gi1 = h0 @ Wih1^T, gh1 = h1 @ Whh1^T ----
    gemm_cols<8, 24>(h0b, H1, fWih1, gbuf,             G2, lane, wv);
    gemm_cols<4, 24>(h1b, H2, fWhh1, gbuf + M_WG * G2, G2, lane, wv);
    __syncthreads();
    for (int i = tid; i < M_WG * H2; i += THREADS) {
      int m = i >> 7, j = i & (H2 - 1);
      const float* gi = gbuf + (size_t)m * G2;
      const float* gh = gbuf + (size_t)M_WG * G2 + (size_t)m * G2;
      float r = sigmoidf_(gi[j]          + bih1[j]          + gh[j]          + bhh1[j]);
      float z = sigmoidf_(gi[j + H2]     + bih1[j + H2]     + gh[j + H2]     + bhh1[j + H2]);
      float n = tanhf(gi[j + 2 * H2] + bih1[j + 2 * H2] +
                      r * (gh[j + 2 * H2] + bhh1[j + 2 * H2]));
      float h = (1.0f - z) * n + z * h1f[i];
      h1f[i] = h; h1b[i] = (__bf16)h;
    }
    __syncthreads();

    // ---- layer 2: gi2 = h1 @ Wih2^T, gh2 = h2 @ Whh2^T ----
    gemm_cols<4, 6>(h1b, H2, fWih2, gbuf,             G3, lane, wv);
    gemm_cols<1, 6>(h2b, H3, fWhh2, gbuf + M_WG * G3, G3, lane, wv);
    __syncthreads();
    for (int i = tid; i < M_WG * H3; i += THREADS) {
      int m = i >> 5, j = i & (H3 - 1);
      const float* gi = gbuf + (size_t)m * G3;
      const float* gh = gbuf + (size_t)M_WG * G3 + (size_t)m * G3;
      float r = sigmoidf_(gi[j]          + bih2[j]          + gh[j]          + bhh2[j]);
      float z = sigmoidf_(gi[j + H3]     + bih2[j + H3]     + gh[j + H3]     + bhh2[j + H3]);
      float n = tanhf(gi[j + 2 * H3] + bih2[j + 2 * H3] +
                      r * (gh[j + 2 * H3] + bhh2[j + 2 * H3]));
      float h = (1.0f - z) * n + z * h2f[i];
      h2f[i] = h; h2b[i] = (__bf16)h;
    }
    __syncthreads();

    // ---- projection: logits = h2 @ Wproj^T (41 padded to 48) ----
    gemm_cols<1, 3>(h2b, H3, fProj, gbuf, NPROJ, lane, wv);
    __syncthreads();

    for (int i = tid; i < M_WG * NVOCAB; i += THREADS) {
      int m = i / NVOCAB, v = i - m * NVOCAB;
      float val = gbuf[(size_t)m * NPROJ + v] + bproj[v];
      outLogits[((size_t)(bm0 + m) * SEQ + t) * NVOCAB + v] = val;
    }
    if (tid < M_WG) {
      int m = tid;
      float best = gbuf[(size_t)m * NPROJ] + bproj[0];
      int bi = 0;
      for (int v = 1; v < NVOCAB; ++v) {
        float val = gbuf[(size_t)m * NPROJ + v] + bproj[v];
        if (val > best) { best = val; bi = v; }   // first-max, matches jnp.argmax
      }
      outPred[(size_t)(bm0 + m) * SEQ + t] = (float)bi;
    }
  }
}

// =====================================================================
// Host launcher
// =====================================================================
extern "C" void kernel_launch(void* const* d_in, const int* in_sizes, int n_in,
                              void* d_out, int out_size, void* d_ws, size_t ws_size,
                              hipStream_t stream) {
  const float* latent = (const float*)d_in[0];
  const int*   tgt    = (const int*)  d_in[1];
  const float* Wemb   = (const float*)d_in[2];
  const float* bemb   = (const float*)d_in[3];
  const float* Winit  = (const float*)d_in[4];
  const float* binit  = (const float*)d_in[5];
  const float* Wih0   = (const float*)d_in[6];
  const float* Whh0   = (const float*)d_in[7];
  const float* bih0   = (const float*)d_in[8];
  const float* bhh0   = (const float*)d_in[9];
  const float* Wih1   = (const float*)d_in[10];
  const float* Whh1   = (const float*)d_in[11];
  const float* bih1   = (const float*)d_in[12];
  const float* bhh1   = (const float*)d_in[13];
  const float* Wih2   = (const float*)d_in[14];
  const float* Whh2   = (const float*)d_in[15];
  const float* bih2   = (const float*)d_in[16];
  const float* bhh2   = (const float*)d_in[17];
  const float* Wproj  = (const float*)d_in[18];
  const float* bproj  = (const float*)d_in[19];

  // ---- workspace carve (256B aligned) ----
  char* ws = (char*)d_ws;
  size_t off = 0;
  auto carve = [&](size_t bytes) -> void* {
    off = (off + 255) & ~(size_t)255;
    void* p = ws + off;
    off += bytes;
    return p;
  };
  __bf16* fWhh0 = (__bf16*)carve(FR_WHH0 * sizeof(__bf16));
  __bf16* fWih1 = (__bf16*)carve(FR_WIH1 * sizeof(__bf16));
  __bf16* fWhh1 = (__bf16*)carve(FR_WHH1 * sizeof(__bf16));
  __bf16* fWih2 = (__bf16*)carve(FR_WIH2 * sizeof(__bf16));
  __bf16* fWhh2 = (__bf16*)carve(FR_WHH2 * sizeof(__bf16));
  __bf16* fProj = (__bf16*)carve(FR_PROJ * sizeof(__bf16));
  float*  gi0t  = (float*)carve((size_t)NVOCAB * G1 * sizeof(float));
  float*  h0i   = (float*)carve((size_t)B_TOT * H1 * sizeof(float));
  float*  h1i   = (float*)carve((size_t)B_TOT * H2 * sizeof(float));
  float*  h2i   = (float*)carve((size_t)B_TOT * H3 * sizeof(float));

  // ---- one-time prep ----
  auto gridFor = [](int total) { return (total + THREADS - 1) / THREADS; };
  pack_frag_kernel<<<gridFor(FR_WHH0), THREADS, 0, stream>>>(Whh0, fWhh0, G1,    H1, 48, 8);
  pack_frag_kernel<<<gridFor(FR_WIH1), THREADS, 0, stream>>>(Wih1, fWih1, G2,    H1, 24, 8);
  pack_frag_kernel<<<gridFor(FR_WHH1), THREADS, 0, stream>>>(Whh1, fWhh1, G2,    H2, 24, 4);
  pack_frag_kernel<<<gridFor(FR_WIH2), THREADS, 0, stream>>>(Wih2, fWih2, G3,    H2,  6, 4);
  pack_frag_kernel<<<gridFor(FR_WHH2), THREADS, 0, stream>>>(Whh2, fWhh2, G3,    H3,  6, 1);
  pack_frag_kernel<<<gridFor(FR_PROJ), THREADS, 0, stream>>>(Wproj, fProj, NVOCAB, H3, 3, 1);
  gi0_table_kernel<<<gridFor(NVOCAB * G1), THREADS, 0, stream>>>(Wemb, bemb, Wih0, bih0, gi0t);
  init_hidden_kernel<<<gridFor(B_TOT * (H1 + H2 + H3)), THREADS, 0, stream>>>(
      latent, Winit, binit, h0i, h1i, h2i);

  // ---- persistent recurrent kernel ----
  float* outLogits = (float*)d_out;
  float* outPred   = outLogits + (size_t)B_TOT * SEQ * NVOCAB;
  size_t ldsBytes =
      (size_t)(M_WG * H1 + M_WG * H2 + M_WG * H3 + M_WG * G1) * sizeof(float) +
      (size_t)(M_WG * H1 + M_WG * H2 + M_WG * H3) * sizeof(__bf16) +
      (size_t)M_WG * sizeof(int);
  gru_persistent_kernel<<<NWG, THREADS, ldsBytes, stream>>>(
      tgt, gi0t, h0i, h1i, h2i,
      fWhh0, fWih1, fWhh1, fWih2, fWhh2, fProj,
      bhh0, bih1, bhh1, bih2, bhh2, bproj,
      outLogits, outPred);
}
